// GCNClassifier_47510928228757
// MI455X (gfx1250) — matrix operand order repaired and checked by reading, not compiled, not run on hardware
//
#include <hip/hip_runtime.h>
#include <hip/hip_bf16.h>

// ---------------------------------------------------------------------------
// GCN classifier for MI455X (gfx1250, wave32, WMMA + TDM).
// Dense GEMMs: TDM (tensor_load_to_lds) stages W and a 64-row A panel into
// LDS (with TDM padding for bank stagger), then 8 waves x 4 row-tiles run
// v_wmma_f32_16x16x32_bf16 out of LDS. SpMM runs as L2-resident gather +
// atomic scatter (feature matrices fit in 192MB L2).
// ---------------------------------------------------------------------------

#define NN   51200      // nodes
#define NE   819200     // edges
#define IND  256        // input feature dim
#define HID  128        // hidden dim
#define BG   64         // graphs

typedef __attribute__((ext_vector_type(16))) __bf16 v16bf;
typedef __attribute__((ext_vector_type(8)))  __bf16 v8bf;
typedef __attribute__((ext_vector_type(8)))  float  v8f;
typedef __attribute__((ext_vector_type(4)))  unsigned int v4u;
typedef __attribute__((ext_vector_type(8)))  int v8i;
typedef __attribute__((ext_vector_type(4)))  int v4i;

// ------------------------------ TDM helpers ---------------------------------

__device__ inline unsigned lds_offset_of(const void* p) {
    // flat LDS aperture keeps the LDS byte offset in addr[31:0]
    return (unsigned)(unsigned long long)p;
}

// 2D tile load: rows_ x k_elems of bf16 (row-major, row stride k_elems) from
// global -> LDS at lds_off, padded +4 DWORDs (8 elems) after every row.
// D# packing per CDNA5 ISA ch.8 (group0 128b, group1 256b; groups 2/3 zero).
template <int K>
__device__ inline void tdm_load_2d_bf16(unsigned lds_off, const void* gsrc,
                                        unsigned rows_) {
    unsigned long long ga = (unsigned long long)gsrc;
    v4u g0;
    g0.x = 0x1u;                                        // count=1, user desc
    g0.y = lds_off;                                     // lds_addr [63:32]
    g0.z = (unsigned)(ga & 0xFFFFFFFFu);                // global_addr lo
    g0.w = (unsigned)((ga >> 32) & 0x1FFFFFFu)          // global_addr [56:32]
         | (2u << 30);                                  // type=2 (image)

    const unsigned td0 = K;                  // tensor_dim0 (elements)
    const unsigned td1 = rows_;              // tensor_dim1
    const unsigned long long s0 = K;         // dim0 stride (elements)
    const unsigned long long s1 = (unsigned long long)K * rows_;
    const unsigned pad_interval = 31 - __builtin_clz((unsigned)K) - 2; // 2^(c+1) DW = K/2 DW
    const unsigned pad_amount   = 3;         // 4 DWORDs = 8 bf16 per row

    v8i g1;
    g1[0] = (int)((1u << 16)                 // data_size = 1 -> 2 bytes
                | (1u << 20)                 // pad_enable
                | (pad_interval << 22)
                | (pad_amount << 25));
    g1[1] = (int)((td0 & 0xFFFFu) << 16);                     // tensor_dim0 lo16
    g1[2] = (int)((td0 >> 16) | ((td1 & 0xFFFFu) << 16));     // td0 hi16 | td1 lo16
    g1[3] = (int)((td1 >> 16) | ((K & 0xFFFFu) << 16));       // td1 hi16 | tile_dim0
    g1[4] = (int)(rows_ & 0xFFFFu);                           // tile_dim1 ; tile_dim2=0
    g1[5] = (int)(s0 & 0xFFFFFFFFu);                          // dim0_stride lo32
    g1[6] = (int)((unsigned)(s0 >> 32) | ((unsigned)(s1 & 0xFFFFu) << 16));
    g1[7] = (int)((s1 >> 16) & 0xFFFFFFFFu);                  // dim1_stride hi32

    v4i g2 = {0, 0, 0, 0};
    v4i g3 = {0, 0, 0, 0};
#if defined(__clang_major__) && (__clang_major__ >= 23)
    v8i g4 = {0, 0, 0, 0, 0, 0, 0, 0};
    __builtin_amdgcn_tensor_load_to_lds(g0, g1, g2, g3, g4, 0);
#else
    __builtin_amdgcn_tensor_load_to_lds(g0, g1, g2, g3, 0);
#endif
}

// ------------------------------ utility kernels -----------------------------

__global__ void zero_f32(float* __restrict__ p, long long n) {
    long long i = (long long)blockIdx.x * blockDim.x + threadIdx.x;
    if (i < n) p[i] = 0.0f;
}

// f32 -> bf16 (round-to-nearest-even), stored as ushort
__global__ void f32_to_bf16(const float* __restrict__ src,
                            unsigned short* __restrict__ dst, long long n) {
    long long i = (long long)blockIdx.x * blockDim.x + threadIdx.x;
    if (i >= n) return;
    unsigned int bits = __float_as_uint(src[i]);
    unsigned int r = bits + 0x7FFFu + ((bits >> 16) & 1u);
    dst[i] = (unsigned short)(r >> 16);
}

// ------------------------------ sparse kernels ------------------------------

__global__ void deg_kernel(const int* __restrict__ row,
                           const float* __restrict__ val,
                           float* __restrict__ deg, int e_cnt) {
    int e = blockIdx.x * blockDim.x + threadIdx.x;
    if (e < e_cnt) atomicAdd(&deg[row[e]], val[e]);
}

__global__ void dinv_kernel(float* __restrict__ deg, int n) {
    int i = blockIdx.x * blockDim.x + threadIdx.x;
    if (i < n) deg[i] = rsqrtf(fmaxf(deg[i], 1e-9f));
}

// one wave per edge; D feature dims split across 32 lanes; atomic scatter-add
template <int D>
__global__ void spmm_kernel(const int* __restrict__ row, const int* __restrict__ col,
                            const float* __restrict__ val, const float* __restrict__ dinv,
                            const float* __restrict__ X, float* __restrict__ AX, int e_cnt) {
    int wid  = (int)((blockIdx.x * blockDim.x + threadIdx.x) >> 5);
    int lane = threadIdx.x & 31;
    if (wid >= e_cnt) return;
    int r = row[wid];
    int c = col[wid];
    float v = val[wid] * dinv[r] * dinv[c];
    const float* xr  = X  + (long long)c * D;
    float*       axr = AX + (long long)r * D;
#pragma unroll
    for (int d = lane; d < D; d += 32) atomicAdd(&axr[d], v * xr[d]);
}

// ------------------------------ WMMA GEMM (TDM-staged) ----------------------
// Y(rows x 128) = [relu]( A(rows x K) * W(128 x K)^T ) [+ addsrc]
// grid.x = rows/64 ; block = 256 (8 waves). Wave w owns HID col-tile w and
// iterates 4 row-tiles. W (128xK) and a 64-row A panel are TDM-staged into
// LDS with +8-element row padding (TDM pad feature) for bank stagger.
// Fragment packing: A/B use contiguous k (lane half c holds k 16c..16c+15); f32 C/D per CDNA5 ISA 7.12.2.
template <int K, bool RELU, bool ADD>
__global__ __launch_bounds__(256)
void wmma_gemm(const unsigned short* __restrict__ A, const unsigned short* __restrict__ W,
               const float* __restrict__ addsrc, float* __restrict__ Y, int rows) {
    (void)rows;
    constexpr int LK = K + 8;                       // padded LDS row stride (elems)
    __shared__ unsigned short sW[HID * LK];
    __shared__ unsigned short sA[64 * LK];

    const int wave = threadIdx.x >> 5;              // 0..7 -> HID col tile
    const int lane = threadIdx.x & 31;
    const int m    = lane & 15;
    const int c    = lane >> 4;                     // half-wave select
    const int row0 = blockIdx.x * 64;
    const int col0 = wave * 16;

    if (wave == 0) {
        tdm_load_2d_bf16<K>(lds_offset_of(sW), W, HID);
        tdm_load_2d_bf16<K>(lds_offset_of(sA), A + (long long)row0 * K, 64);
        __builtin_amdgcn_s_wait_tensorcnt(0);
    }
    __syncthreads();

    v8f acc[4] = {};
    const unsigned short* brow = &sW[(col0 + m) * LK];
#pragma unroll
    for (int k0 = 0; k0 < K; k0 += 32) {
        // B lane n+16c holds K[k0+16c .. +16) of column n (two 16B LDS loads)
        v8bf blo = *(const v8bf*)(brow + k0 + 16 * c);
        v8bf bhi = *(const v8bf*)(brow + k0 + 16 * c + 8);
        v16bf b = __builtin_shufflevector(blo, bhi,
                    0, 1, 2, 3, 4, 5, 6, 7, 8, 9, 10, 11, 12, 13, 14, 15);
#pragma unroll
        for (int rt = 0; rt < 4; ++rt) {
            const unsigned short* arow = &sA[(rt * 16 + m) * LK + k0];
            // A lane m+16c: elems 0..15 = K[k0+16c..+16)
            v8bf alo = *(const v8bf*)(arow + 16 * c);
            v8bf ahi = *(const v8bf*)(arow + 16 * c + 8);
            v16bf a = __builtin_shufflevector(alo, ahi,
                        0, 1, 2, 3, 4, 5, 6, 7, 8, 9, 10, 11, 12, 13, 14, 15);
            acc[rt] = __builtin_amdgcn_wmma_f32_16x16x32_bf16(
                          false, a, false, b, (short)0, acc[rt], false, false);
        }
    }

    // D lane n+16c, VGPR r holds D[8c+r][n]
    const int n = lane & 15;
#pragma unroll
    for (int rt = 0; rt < 4; ++rt) {
        long long base = (long long)(row0 + rt * 16 + 8 * c) * HID + col0 + n;
#pragma unroll
        for (int r = 0; r < 8; ++r) {
            float v = acc[rt][r];
            if (RELU) v = fmaxf(v, 0.0f);
            if (ADD)  v += addsrc[base + (long long)r * HID];
            Y[base + (long long)r * HID] = v;
        }
    }
}

// ------------------------------ epilogue kernels ----------------------------

// LayerNorm over 128 features; one wave per node (4 values/lane, shfl_xor tree)
__global__ void ln_kernel(float* __restrict__ h, const float* __restrict__ gamma,
                          const float* __restrict__ beta, int n) {
    int wid  = (int)((blockIdx.x * blockDim.x + threadIdx.x) >> 5);
    int lane = threadIdx.x & 31;
    if (wid >= n) return;
    float* hp = h + (long long)wid * HID;
    float x[4];
#pragma unroll
    for (int j = 0; j < 4; ++j) x[j] = hp[lane + 32 * j];
    float s = x[0] + x[1] + x[2] + x[3];
#pragma unroll
    for (int o = 16; o > 0; o >>= 1) s += __shfl_xor(s, o, 32);
    float mu = s * (1.0f / 128.0f);
    float q = 0.0f;
#pragma unroll
    for (int j = 0; j < 4; ++j) { float d = x[j] - mu; q += d * d; }
#pragma unroll
    for (int o = 16; o > 0; o >>= 1) q += __shfl_xor(q, o, 32);
    float inv = rsqrtf(q * (1.0f / 128.0f) + 1e-5f);
#pragma unroll
    for (int j = 0; j < 4; ++j) {
        int f = lane + 32 * j;
        hp[f] = (x[j] - mu) * inv * gamma[f] + beta[f];
    }
}

// one block per graph, one thread per feature; contiguous node segments
__global__ void pool_kernel(const float* __restrict__ h, const int* __restrict__ ptr,
                            float* __restrict__ H) {
    int g = blockIdx.x;
    int f = threadIdx.x;            // 0..127
    int s = ptr[g], e = ptr[g + 1];
    float sum = 0.0f, mx = -3.4028235e38f;
    for (int i = s; i < e; ++i) {
        float v = h[(long long)i * HID + f];
        sum += v;
        mx = fmaxf(mx, v);
    }
    int cnt = e - s;
    float mean = sum / fmaxf((float)cnt, 1.0f);
    if (cnt <= 0) { mean = 0.0f; mx = 0.0f; }
    H[g * (2 * HID) + f]       = mean;
    H[g * (2 * HID) + HID + f] = mx;
}

// out(64x2) = H(64x256) @ Wcls(2x256)^T + b
__global__ void cls_kernel(const float* __restrict__ H, const float* __restrict__ Wcls,
                           const float* __restrict__ b, float* __restrict__ out) {
    int idx = blockIdx.x * blockDim.x + threadIdx.x;   // g*2 + o
    if (idx >= BG * 2) return;
    int g = idx >> 1, o = idx & 1;
    const float* hr = H + g * (2 * HID);
    const float* wr = Wcls + o * (2 * HID);
    float acc = b[o];
    for (int k = 0; k < 2 * HID; ++k) acc += hr[k] * wr[k];
    out[idx] = acc;
}

// ------------------------------ launcher ------------------------------------

extern "C" void kernel_launch(void* const* d_in, const int* in_sizes, int n_in,
                              void* d_out, int out_size, void* d_ws, size_t ws_size,
                              hipStream_t stream) {
    (void)in_sizes; (void)n_in; (void)out_size; (void)ws_size;
    const float* X     = (const float*)d_in[0];
    const int*   eidx  = (const int*)d_in[1];      // (2, NE): row then col
    const float* eval_ = (const float*)d_in[2];
    const int*   ptr   = (const int*)d_in[3];
    const float* W1    = (const float*)d_in[4];
    const float* W2    = (const float*)d_in[5];
    const float* Wres  = (const float*)d_in[6];
    const float* gam   = (const float*)d_in[7];
    const float* bet   = (const float*)d_in[8];
    const float* Wcls  = (const float*)d_in[9];
    const float* bcls  = (const float*)d_in[10];
    float* out = (float*)d_out;

    const int* row = eidx;
    const int* col = eidx + NE;

    // workspace carve-up (256B aligned)
    char* w = (char*)d_ws;
    size_t off = 0;
    auto carve = [&](size_t bytes) -> void* {
        void* p = w + off;
        off = (off + bytes + 255) & ~(size_t)255;
        return p;
    };
    float*          dinv   = (float*)carve((size_t)NN * 4);
    unsigned short* Xbf    = (unsigned short*)carve((size_t)NN * IND * 2);
    float*          AX     = (float*)carve((size_t)NN * IND * 4);
    unsigned short* AXbf   = (unsigned short*)carve((size_t)NN * IND * 2);
    float*          H1     = (float*)carve((size_t)NN * HID * 4);
    float*          AX2    = (float*)carve((size_t)NN * HID * 4);
    unsigned short* AX2bf  = (unsigned short*)carve((size_t)NN * HID * 2);
    float*          RES    = (float*)carve((size_t)NN * HID * 4);
    float*          Hp     = (float*)carve((size_t)BG * 2 * HID * 4);
    unsigned short* W1bf   = (unsigned short*)carve((size_t)HID * IND * 2);
    unsigned short* W2bf   = (unsigned short*)carve((size_t)HID * HID * 2);
    unsigned short* Wrbf   = (unsigned short*)carve((size_t)HID * IND * 2);

    const int BT = 256;
    auto blocks = [](long long n, int bt) { return (unsigned)((n + bt - 1) / bt); };

    // 1) degree -> d^-1/2
    zero_f32<<<blocks(NN, BT), BT, 0, stream>>>(dinv, NN);
    deg_kernel<<<blocks(NE, BT), BT, 0, stream>>>(row, eval_, dinv, NE);
    dinv_kernel<<<blocks(NN, BT), BT, 0, stream>>>(dinv, NN);

    // 2) bf16 casts of static operands
    f32_to_bf16<<<blocks((long long)NN * IND, BT), BT, 0, stream>>>(X, Xbf, (long long)NN * IND);
    f32_to_bf16<<<blocks((long long)HID * IND, BT), BT, 0, stream>>>(W1, W1bf, (long long)HID * IND);
    f32_to_bf16<<<blocks((long long)HID * HID, BT), BT, 0, stream>>>(W2, W2bf, (long long)HID * HID);
    f32_to_bf16<<<blocks((long long)HID * IND, BT), BT, 0, stream>>>(Wres, Wrbf, (long long)HID * IND);

    // 3) SpMM1: AX = A_norm * X   (atomic scatter, L2-resident)
    zero_f32<<<blocks((long long)NN * IND, BT), BT, 0, stream>>>(AX, (long long)NN * IND);
    spmm_kernel<IND><<<blocks((long long)NE * 32, BT), BT, 0, stream>>>(row, col, eval_, dinv, X, AX, NE);
    f32_to_bf16<<<blocks((long long)NN * IND, BT), BT, 0, stream>>>(AX, AXbf, (long long)NN * IND);

    // 4) H1 = relu(AX @ W1^T)  -- TDM + WMMA
    wmma_gemm<IND, true, false><<<NN / 64, 256, 0, stream>>>(AXbf, W1bf, nullptr, H1, NN);

    // 5) SpMM2: AX2 = A_norm * H1
    zero_f32<<<blocks((long long)NN * HID, BT), BT, 0, stream>>>(AX2, (long long)NN * HID);
    spmm_kernel<HID><<<blocks((long long)NE * 32, BT), BT, 0, stream>>>(row, col, eval_, dinv, H1, AX2, NE);
    f32_to_bf16<<<blocks((long long)NN * HID, BT), BT, 0, stream>>>(AX2, AX2bf, (long long)NN * HID);

    // 6) RES = X @ Wres^T  -- TDM + WMMA
    wmma_gemm<IND, false, false><<<NN / 64, 256, 0, stream>>>(Xbf, Wrbf, nullptr, RES, NN);

    // 7) RES = relu(AX2 @ W2^T) + RES  -- TDM + WMMA, fused epilogue, in place
    wmma_gemm<HID, true, true><<<NN / 64, 256, 0, stream>>>(AX2bf, W2bf, RES, RES, NN);

    // 8) LayerNorm (in place), pooling, classifier
    ln_kernel<<<blocks((long long)NN * 32, BT), BT, 0, stream>>>(RES, gam, bet, NN);
    pool_kernel<<<BG, HID, 0, stream>>>(RES, ptr, Hp);
    cls_kernel<<<1, 128, 0, stream>>>(Hp, Wcls, bcls, out);
}
